// Aggregate_31645319037546
// MI455X (gfx1250) — compile-verified, matrix operand order checked by reading, not run on hardware
//
#include <hip/hip_runtime.h>
#include <hip/hip_bf16.h>
#include <cstdint>

typedef _Float16 half_t;
typedef __attribute__((ext_vector_type(16))) _Float16 v16h;
typedef __attribute__((ext_vector_type(8)))  float    v8f;
typedef uint32_t u32x4 __attribute__((ext_vector_type(4)));
typedef uint32_t u32x8 __attribute__((ext_vector_type(8)));

__device__ __forceinline__ float lrelu_f(float v) { return v > 0.f ? v : 0.1f * v; }

// LDS byte offset from a generic pointer: LDS aperture keeps offset in bits[31:0]
__device__ __forceinline__ uint32_t lds_off_u32(const void* p) {
    return (uint32_t)(uintptr_t)p;
}

// ---------------------------------------------------------------------------
// f32 -> f16 weight conversion with zero padding to (Cpad x Kpad)
// ---------------------------------------------------------------------------
__global__ void k_cvt_pad(const float* __restrict__ in, half_t* __restrict__ out,
                          int Cout, int K, int Kpad, int Cpad)
{
    int i = blockIdx.x * blockDim.x + threadIdx.x;
    int total = Cpad * Kpad;
    if (i >= total) return;
    int co = i / Kpad, k = i - co * Kpad;
    out[i] = (co < Cout && k < K) ? (half_t)in[(size_t)co * K + k] : (half_t)0.f;
}

__global__ void k_copy(const float* __restrict__ in, float* __restrict__ out, int n) {
    int i = blockIdx.x * blockDim.x + threadIdx.x;
    if (i < n) out[i] = in[i];
}

// ---------------------------------------------------------------------------
// Generic NCHW conv via implicit GEMM + WMMA f16 (f32 accumulate).
// Block: 256 threads = 8 wave32. Tile: 128 pixels (M) x 32 out-channels (N).
// A tile: im2col staged by VALU (f32->f16 + border zeros).
// B tile: ASYNC global->LDS copy (weights padded, guard-free).
// ---------------------------------------------------------------------------
__global__ void k_conv(const float* __restrict__ in, const half_t* __restrict__ wf,
                       const float* __restrict__ bias, float* __restrict__ out,
                       int Cin, int H, int W, int Cout, int Kpad,
                       int kh, int kw, int ph, int pw, int dolrelu)
{
    __shared__ half_t As[128 * 32];
    __shared__ half_t Bs[32 * 32];
    const int HW  = H * W;
    const int khw = kh * kw;
    const int K   = Cin * khw;
    const int m0  = blockIdx.x * 128;
    const int n0  = blockIdx.y * 32;
    const int tid = threadIdx.x;
    const int lane = tid & 31;
    const int wave = tid >> 5;
    const int khalf = (lane < 16) ? 0 : 8;

    v8f acc0 = {}, acc1 = {};
    for (int k0 = 0; k0 < K; k0 += 32) {
        // ---- stage A (im2col, f32 -> f16, zero borders) ----
        for (int idx = tid; idx < 128 * 32; idx += 256) {
            int ml = idx >> 5, kk = idx & 31;
            int k = k0 + kk;
            int m = m0 + ml;
            half_t v = (half_t)0.f;
            if (k < K && m < HW) {
                int c = k / khw; int rem = k - c * khw;
                int ki = rem / kw, kj = rem - ki * kw;
                int y = (m / W) + ki - ph;
                int x = (m % W) + kj - pw;
                if ((unsigned)y < (unsigned)H && (unsigned)x < (unsigned)W)
                    v = (half_t)in[c * HW + y * W + x];
            }
            As[idx] = v;
        }
        // ---- stage B: async global->LDS (2KB tile, one b64 per lane) ----
        {
            int row = tid >> 3;      // 0..31 out-channel row
            int chunk = tid & 7;     // 8 bytes each
            uint64_t gp = (uint64_t)(uintptr_t)(wf + (size_t)(n0 + row) * Kpad + k0 + chunk * 4);
            uint32_t lp = lds_off_u32(&Bs[row * 32 + chunk * 4]);
            asm volatile("global_load_async_to_lds_b64 %0, %1, off"
                         :: "v"(lp), "v"(gp) : "memory");
        }
        if (k0 + 32 < Kpad) __builtin_prefetch(&wf[(size_t)n0 * Kpad + k0 + 32], 0, 1);
        asm volatile("s_wait_asynccnt 0x0" ::: "memory");
        __syncthreads();

        // ---- build fragments (16-bit A layout; half2 dword loads) ----
        const int mrow = (wave << 4) + (lane & 15);
        union { v16h v; uint32_t u[8]; } Af, Bf0, Bf1;
        const uint32_t* ap  = (const uint32_t*)&As[mrow * 32];
        const uint32_t* bp0 = (const uint32_t*)&Bs[(lane & 15) * 32];
        const uint32_t* bp1 = (const uint32_t*)&Bs[(16 + (lane & 15)) * 32];
        #pragma unroll
        for (int e2 = 0; e2 < 8; ++e2) {
            int Kb = ((e2 < 4) ? (2 * e2) : (2 * e2 + 8)) + khalf;  // even
            Af.u[e2]  = ap[Kb >> 1];
            Bf0.u[e2] = bp0[Kb >> 1];
            Bf1.u[e2] = bp1[Kb >> 1];
        }
        acc0 = __builtin_amdgcn_wmma_f32_16x16x32_f16(false, Af.v, false, Bf0.v,
                                                      (short)0, acc0, false, false);
        acc1 = __builtin_amdgcn_wmma_f32_16x16x32_f16(false, Af.v, false, Bf1.v,
                                                      (short)0, acc1, false, false);
        __syncthreads();
    }

    // ---- epilogue: bias + optional leaky relu ----
    #pragma unroll
    for (int nt = 0; nt < 2; ++nt) {
        int co = n0 + nt * 16 + (lane & 15);
        if (co < Cout) {
            float b = bias[co];
            const v8f& acc = nt ? acc1 : acc0;
            #pragma unroll
            for (int r = 0; r < 8; ++r) {
                int m = m0 + (wave << 4) + r + khalf;
                if (m < HW) {
                    float v = acc[r] + b;
                    if (dolrelu) v = lrelu_f(v);
                    out[(size_t)co * HW + m] = v;
                }
            }
        }
    }
}

// ---------------------------------------------------------------------------
// L2 normalization over 64 channels per pixel
// ---------------------------------------------------------------------------
__global__ void k_l2norm(const float* __restrict__ in, float* __restrict__ out,
                         int C, int HW)
{
    int p = blockIdx.x * blockDim.x + threadIdx.x;
    if (p >= HW) return;
    float s = 0.f;
    for (int c = 0; c < C; ++c) { float v = in[c * HW + p]; s += v * v; }
    float inv = 1.f / fmaxf(sqrtf(s), 1e-12f);
    for (int c = 0; c < C; ++c) out[c * HW + p] = in[c * HW + p] * inv;
}

// ---------------------------------------------------------------------------
// Correlation (3x3 kernel window) + top-4 displacement indices
// ---------------------------------------------------------------------------
__global__ void k_corr_topk(const float* __restrict__ f1n, const float* __restrict__ f2n,
                            int* __restrict__ ind, int H, int W, int patch)
{
    int p = blockIdx.x * blockDim.x + threadIdx.x;
    const int HW = H * W;
    if (p >= HW) return;
    const int y = p / W, x = p % W;
    const int ps = (patch - 1) / 2;
    float bv[4] = {-1e30f, -1e30f, -1e30f, -1e30f};
    int   bi[4] = {0, 0, 0, 0};
    const int P2 = patch * patch;
    for (int d = 0; d < P2; ++d) {
        int dh = d / patch, dw = d % patch;
        float s = 0.f;
        for (int ki = 0; ki < 3; ++ki) {
            for (int kj = 0; kj < 3; ++kj) {
                int ay = y + ki - 1, ax = x + kj - 1;
                int by = ay + dh - ps, bx = ax + dw - ps;
                if ((unsigned)ay < (unsigned)H && (unsigned)ax < (unsigned)W &&
                    (unsigned)by < (unsigned)H && (unsigned)bx < (unsigned)W) {
                    const float* a = f1n + ay * W + ax;
                    const float* b = f2n + by * W + bx;
                    float acc = 0.f;
                    for (int c = 0; c < 64; ++c) acc = fmaf(a[c * HW], b[c * HW], acc);
                    s += acc;
                }
            }
        }
        if (s > bv[3]) {                       // insertion top-4 (stable: earlier d wins ties)
            int j = 3;
            while (j > 0 && s > bv[j - 1]) { bv[j] = bv[j - 1]; bi[j] = bi[j - 1]; --j; }
            bv[j] = s; bi[j] = d;
        }
    }
    ind[p * 4 + 0] = bi[0]; ind[p * 4 + 1] = bi[1];
    ind[p * 4 + 2] = bi[2]; ind[p * 4 + 3] = bi[3];
}

// ---------------------------------------------------------------------------
// 2x bilinear upsample (half-pixel centers, edge clamp)
// ---------------------------------------------------------------------------
__global__ void k_up2(const float* __restrict__ in, float* __restrict__ out,
                      int C, int H, int W)
{
    const int oH = 2 * H, oW = 2 * W;
    int total = C * oH * oW;
    int i = blockIdx.x * blockDim.x + threadIdx.x;
    if (i >= total) return;
    int ox = i % oW; int t = i / oW; int oy = t % oH; int c = t / oH;
    float sy = 0.5f * oy - 0.25f;
    float sx = 0.5f * ox - 0.25f;
    int y0 = (int)floorf(sy); float fy = sy - y0;
    int x0 = (int)floorf(sx); float fx = sx - x0;
    int y1 = min(max(y0 + 1, 0), H - 1); y0 = min(max(y0, 0), H - 1);
    int x1 = min(max(x0 + 1, 0), W - 1); x0 = min(max(x0, 0), W - 1);
    const float* pch = in + (size_t)c * H * W;
    float v = (1.f - fy) * ((1.f - fx) * pch[y0 * W + x0] + fx * pch[y0 * W + x1]) +
              fy         * ((1.f - fx) * pch[y1 * W + x0] + fx * pch[y1 * W + x1]);
    out[i] = v;
}

// ---------------------------------------------------------------------------
// Fused KNN tail: patch gather + avg*am concat + 320->64 3x3 conv (WMMA GEMM,
// K = 2880) + bias/lrelu + mask-weighted 9-tap reduction + final lrelu.
// One wave per pixel; 8 pixels per block. Weight tiles (64 rows x 32 halves,
// row stride 2880) fetched by the Tensor Data Mover into LDS.
// ---------------------------------------------------------------------------
__global__ void k_knn_tail(const float* __restrict__ nbrf, const int* __restrict__ ind,
                           const float* __restrict__ am, const float* __restrict__ maskf,
                           const half_t* __restrict__ wnn, const float* __restrict__ bnn,
                           float* __restrict__ out, int H, int W, int patch, int ps)
{
    __shared__ half_t pt[8][2880];     // 320 ch x 3x3 gathered patch per pixel (f16)
    __shared__ half_t Bs[64 * 32];     // weight tile: [oc][kk]
    const int HW   = H * W;
    const int tid  = threadIdx.x;
    const int lane = tid & 31;
    const int wave = tid >> 5;
    const int pix  = blockIdx.x * 8 + wave;
    const int y = pix / W, x = pix % W;

    // ---- phase 1: gather 4 neighbor patches (cc < 256) ----
    int idxs[4];
    #pragma unroll
    for (int q = 0; q < 4; ++q) idxs[q] = ind[pix * 4 + q];
    for (int e = lane; e < 2304; e += 32) {
        int cc = e / 9, t = e - cc * 9;
        int nb = cc >> 6, c = cc & 63;
        int d = idxs[nb];
        int dh = d / patch, dw = d - dh * patch;
        int r = t / 3, s = t - r * 3;
        int row = y + dh + r - ps, col = x + dw + s - ps;
        half_t v = (half_t)0.f;
        if ((unsigned)row < (unsigned)H && (unsigned)col < (unsigned)W)
            v = (half_t)nbrf[c * HW + row * W + col];
        pt[wave][e] = v;
    }
    __syncthreads();
    // ---- avg * am block (cc in [256,320)) ----
    for (int e = lane; e < 576; e += 32) {
        int c = e / 9, t = e - c * 9;
        float a = 0.25f * ((float)pt[wave][(c)       * 9 + t] +
                           (float)pt[wave][(64 + c)  * 9 + t] +
                           (float)pt[wave][(128 + c) * 9 + t] +
                           (float)pt[wave][(192 + c) * 9 + t]);
        pt[wave][2304 + e] = (half_t)(a * am[t * HW + pix]);
    }

    // TDM descriptor group 1 (constant across k-steps):
    //  wg_mask=0, data_size=1 (2B), tensor_dim0=2880, tensor_dim1=64,
    //  tile_dim0=32, tile_dim1=64, tensor_dim0_stride=2880
    const u32x8 g1 = { 0x00010000u,            // data_size=2B
                       (2880u & 0xFFFFu) << 16,  // tensor_dim0[15:0]
                       (64u << 16),              // dim0 hi=0 | tensor_dim1 lo
                       (32u << 16),              // dim1 hi=0 | tile_dim0
                       64u,                      // tile_dim1 | tile_dim2=0
                       2880u,                    // tensor_dim0_stride[31:0]
                       0u, 0u };
    const uint32_t bs_lds = lds_off_u32(&Bs[0]);

    // ---- phase 2: GEMM  M=9(pad 16) x N=64 x K=2880 ----
    v8f accs[4] = {{}, {}, {}, {}};
    const int j  = lane & 15;          // output position row (valid if j < 9)
    const int pj = j / 3, qj = j - pj * 3;
    const int khalf = (lane < 16) ? 0 : 8;
    for (int k0 = 0; k0 < 2880; k0 += 32) {
        __syncthreads();               // previous tile fully consumed
        if (wave == 0) {
            uint64_t ga = (uint64_t)(uintptr_t)wnn + (uint64_t)k0 * 2u;
            u32x4 g0 = { 1u,                               // count=1
                         bs_lds,                           // lds_addr
                         (uint32_t)ga,                     // global_addr lo
                         (uint32_t)(ga >> 32) | (2u << 30) // global_addr hi | type=2
                       };
            asm volatile("tensor_load_to_lds %0, %1" :: "s"(g0), "s"(g1) : "memory");
            __builtin_amdgcn_s_wait_tensorcnt(0);
        }
        __syncthreads();               // tile published to all waves

        // A fragment: im2col over the 3x3 patch (pad 1), replicated from LDS
        union { v16h v; half_t h[16]; } Af;
        #pragma unroll
        for (int e = 0; e < 16; ++e) {
            int K = ((e < 8) ? e : e + 8) + khalf;
            int k = k0 + K;
            int cc = k / 9, t = k - cc * 9;
            int ki = t / 3, kj = t - ki * 3;
            int r = pj + ki - 1, s = qj + kj - 1;
            half_t v = (half_t)0.f;
            if (j < 9 && (unsigned)r < 3u && (unsigned)s < 3u)
                v = pt[wave][cc * 9 + r * 3 + s];
            Af.h[e] = v;
        }
        #pragma unroll
        for (int nt = 0; nt < 4; ++nt) {
            union { v16h v; uint32_t u[8]; } Bf;
            const uint32_t* bp = (const uint32_t*)&Bs[(nt * 16 + j) * 32];
            #pragma unroll
            for (int e2 = 0; e2 < 8; ++e2) {
                int Kb = ((e2 < 4) ? (2 * e2) : (2 * e2 + 8)) + khalf;
                Bf.u[e2] = bp[Kb >> 1];
            }
            accs[nt] = __builtin_amdgcn_wmma_f32_16x16x32_f16(false, Af.v, false, Bf.v,
                                                              (short)0, accs[nt], false, false);
        }
    }

    // ---- epilogue: x = lrelu(acc+b); out[co] = lrelu( sum_j x[j]*mask[g*9+j] ) ----
    const int ocl = lane & 15;
    #pragma unroll
    for (int nt = 0; nt < 4; ++nt) {
        int co = nt * 16 + ocl;
        float b = bnn[co];
        int g = co >> 3;
        float partial = 0.f;
        #pragma unroll
        for (int r = 0; r < 8; ++r) {
            int jj = r + khalf;
            if (jj < 9) {
                float v = lrelu_f(accs[nt][r] + b);
                partial += v * maskf[(g * 9 + jj) * HW + pix];
            }
        }
        float tot = partial + __shfl_xor(partial, 16, 32);
        if (lane < 16) out[(size_t)co * HW + pix] = lrelu_f(tot);
    }
}

// ---------------------------------------------------------------------------
// Host side
// ---------------------------------------------------------------------------
static const int SPEC[28][4] = {
    {64,128,3,3},{64,64,3,3},{64,64,7,1},{64,64,1,7},{72,64,3,3},{64,3,3,3},{9,64,1,1},{64,320,3,3},
    {64,128,3,3},{64,128,3,3},{64,64,3,3},{64,64,7,1},{64,64,1,7},{64,128,3,3},{72,64,3,3},{64,3,3,3},{9,64,1,1},{64,320,3,3},
    {64,128,3,3},{64,128,3,3},{64,64,3,3},{64,64,7,1},{64,64,1,7},{64,128,3,3},{72,64,3,3},{64,3,3,3},{9,64,1,1},{64,320,3,3}
};

extern "C" void kernel_launch(void* const* d_in, const int* in_sizes, int n_in,
                              void* d_out, int out_size, void* d_ws, size_t ws_size,
                              hipStream_t stream)
{
    (void)in_sizes; (void)n_in; (void)out_size; (void)ws_size;
    char* ws = (char*)d_ws;
    size_t off = 0;
    auto alloc = [&](size_t bytes) -> char* {
        char* p = ws + off;
        off += (bytes + 255) & ~(size_t)255;
        return p;
    };

    half_t* wf[28];
    int Kp[28], Cp[28];
    for (int s = 0; s < 28; ++s) {
        int K = SPEC[s][1] * SPEC[s][2] * SPEC[s][3];
        Kp[s] = (K + 31) & ~31;
        Cp[s] = (SPEC[s][0] + 31) & ~31;
        wf[s] = (half_t*)alloc((size_t)Cp[s] * Kp[s] * sizeof(half_t));
    }
    const size_t MHW = 128 * 128;
    float* bufA  = (float*)alloc(128 * MHW * 4);
    float* bufB  = (float*)alloc(128 * MHW * 4);
    float* bufC  = (float*)alloc(64 * MHW * 4);
    float* wfe3  = (float*)alloc(64 * 32 * 32 * 4);
    float* wfe2  = (float*)alloc(64 * 64 * 64 * 4);
    float* wfe1  = (float*)alloc(64 * MHW * 4);
    float* f1n   = (float*)alloc(64 * MHW * 4);
    float* f2n   = (float*)alloc(64 * MHW * 4);
    int*   indb  = (int*)alloc(MHW * 4 * sizeof(int));
    float* amb   = (float*)alloc(9 * MHW * 4);
    float* maskb = (float*)alloc(72 * MHW * 4);
    float* knnb  = (float*)alloc(64 * MHW * 4);
    float* L3o   = (float*)alloc(64 * 32 * 32 * 4);
    float* L2o   = (float*)alloc(64 * 64 * 64 * 4);

    // convert all conv weights to padded f16 once per launch (deterministic)
    for (int s = 0; s < 28; ++s) {
        int K = SPEC[s][1] * SPEC[s][2] * SPEC[s][3];
        int n = Cp[s] * Kp[s];
        k_cvt_pad<<<(n + 255) / 256, 256, 0, stream>>>((const float*)d_in[9 + 2 * s],
                                                       wf[s], SPEC[s][0], K, Kp[s], Cp[s]);
    }
    auto Bias = [&](int s) { return (const float*)d_in[9 + 2 * s + 1]; };
    auto conv = [&](const float* in, int Cin, int H, int Wd, int s, float* out,
                    int kh, int kw, int ph, int pw, int lr) {
        int Cout = SPEC[s][0];
        int HW = H * Wd;
        dim3 g((HW + 127) / 128, (Cout + 31) / 32);
        k_conv<<<g, 256, 0, stream>>>(in, wf[s], Bias(s), out, Cin, H, Wd, Cout, Kp[s],
                                      kh, kw, ph, pw, lr);
    };
    auto copy = [&](const float* src, float* dst, int n) {
        k_copy<<<(n + 255) / 256, 256, 0, stream>>>(src, dst, n);
    };

    // ================= L3 (32x32, patch 7, ps 4) =================
    {
        int H = 32, Wd = 32, HW = H * Wd;
        const float* nbr = (const float*)d_in[2];
        const float* ref = (const float*)d_in[5];
        const float* rec = (const float*)d_in[8];
        copy(nbr, bufA, 64 * HW); copy(ref, bufA + 64 * HW, 64 * HW);
        conv(bufA, 128, H, Wd, 0, bufB, 3, 3, 1, 1, 1);
        conv(bufB, 64, H, Wd, 1, bufA, 3, 3, 1, 1, 1);
        conv(bufA, 64, H, Wd, 2, bufB, 7, 1, 3, 0, 1);
        conv(bufB, 64, H, Wd, 3, wfe3, 1, 7, 0, 3, 1);
        conv(wfe3, 64, H, Wd, 4, maskb, 3, 3, 1, 1, 0);
        k_l2norm<<<(HW + 255) / 256, 256, 0, stream>>>(ref, f1n, 64, HW);
        k_l2norm<<<(HW + 255) / 256, 256, 0, stream>>>(nbr, f2n, 64, HW);
        k_corr_topk<<<(HW + 255) / 256, 256, 0, stream>>>(f1n, f2n, indb, H, Wd, 7);
        conv(rec, 3, H, Wd, 5, bufB, 3, 3, 1, 1, 1);
        conv(bufB, 64, H, Wd, 6, amb, 1, 1, 0, 0, 1);
        k_knn_tail<<<HW / 8, 256, 0, stream>>>(nbr, indb, amb, maskb, wf[7], Bias(7),
                                               L3o, H, Wd, 7, 4);
    }
    // ================= L2 (64x64, patch 11, ps 6) =================
    {
        int H = 64, Wd = 64, HW = H * Wd;
        const float* nbr = (const float*)d_in[1];
        const float* ref = (const float*)d_in[4];
        const float* rec = (const float*)d_in[7];
        copy(nbr, bufA, 64 * HW); copy(ref, bufA + 64 * HW, 64 * HW);
        conv(bufA, 128, H, Wd, 8, bufB, 3, 3, 1, 1, 1);
        k_up2<<<(64 * HW + 255) / 256, 256, 0, stream>>>(wfe3, bufC, 64, 32, 32);
        copy(bufB, bufA, 64 * HW); copy(bufC, bufA + 64 * HW, 64 * HW);
        conv(bufA, 128, H, Wd, 9, bufB, 3, 3, 1, 1, 1);
        conv(bufB, 64, H, Wd, 10, bufA, 3, 3, 1, 1, 1);
        conv(bufA, 64, H, Wd, 11, bufB, 7, 1, 3, 0, 1);
        conv(bufB, 64, H, Wd, 12, wfe2, 1, 7, 0, 3, 1);
        conv(wfe2, 64, H, Wd, 14, maskb, 3, 3, 1, 1, 0);
        k_l2norm<<<(HW + 255) / 256, 256, 0, stream>>>(ref, f1n, 64, HW);
        k_l2norm<<<(HW + 255) / 256, 256, 0, stream>>>(nbr, f2n, 64, HW);
        k_corr_topk<<<(HW + 255) / 256, 256, 0, stream>>>(f1n, f2n, indb, H, Wd, 11);
        conv(rec, 3, H, Wd, 15, bufB, 3, 3, 1, 1, 1);
        conv(bufB, 64, H, Wd, 16, amb, 1, 1, 0, 0, 1);
        k_knn_tail<<<HW / 8, 256, 0, stream>>>(nbr, indb, amb, maskb, wf[17], Bias(17),
                                               knnb, H, Wd, 11, 6);
        k_up2<<<(64 * HW + 255) / 256, 256, 0, stream>>>(L3o, bufC, 64, 32, 32);
        copy(knnb, bufA, 64 * HW); copy(bufC, bufA + 64 * HW, 64 * HW);
        conv(bufA, 128, H, Wd, 13, L2o, 3, 3, 1, 1, 1);
    }
    // ================= L1 (128x128, patch 15, ps 8) =================
    {
        int H = 128, Wd = 128, HW = H * Wd;
        const float* nbr = (const float*)d_in[0];
        const float* ref = (const float*)d_in[3];
        const float* rec = (const float*)d_in[6];
        copy(nbr, bufA, 64 * HW); copy(ref, bufA + 64 * HW, 64 * HW);
        conv(bufA, 128, H, Wd, 18, bufB, 3, 3, 1, 1, 1);
        k_up2<<<(64 * HW + 255) / 256, 256, 0, stream>>>(wfe2, bufC, 64, 64, 64);
        copy(bufB, bufA, 64 * HW); copy(bufC, bufA + 64 * HW, 64 * HW);
        conv(bufA, 128, H, Wd, 19, bufB, 3, 3, 1, 1, 1);
        conv(bufB, 64, H, Wd, 20, bufA, 3, 3, 1, 1, 1);
        conv(bufA, 64, H, Wd, 21, bufB, 7, 1, 3, 0, 1);
        conv(bufB, 64, H, Wd, 22, wfe1, 1, 7, 0, 3, 1);
        conv(wfe1, 64, H, Wd, 24, maskb, 3, 3, 1, 1, 0);
        k_l2norm<<<(HW + 255) / 256, 256, 0, stream>>>(ref, f1n, 64, HW);
        k_l2norm<<<(HW + 255) / 256, 256, 0, stream>>>(nbr, f2n, 64, HW);
        k_corr_topk<<<(HW + 255) / 256, 256, 0, stream>>>(f1n, f2n, indb, H, Wd, 15);
        conv(rec, 3, H, Wd, 25, bufB, 3, 3, 1, 1, 1);
        conv(bufB, 64, H, Wd, 26, amb, 1, 1, 0, 0, 1);
        k_knn_tail<<<HW / 8, 256, 0, stream>>>(nbr, indb, amb, maskb, wf[27], Bias(27),
                                               knnb, H, Wd, 15, 8);
        k_up2<<<(64 * HW + 255) / 256, 256, 0, stream>>>(L2o, bufC, 64, 64, 64);
        copy(knnb, bufA, 64 * HW); copy(bufC, bufA + 64 * HW, 64 * HW);
        conv(bufA, 128, H, Wd, 23, (float*)d_out, 3, 3, 1, 1, 1);
    }
}
